// FeedForwardQuantum_65481071402935
// MI455X (gfx1250) — compile-verified, matrix operand order checked by reading, not run on hardware
//
#include <hip/hip_runtime.h>

typedef _Float16 half8 __attribute__((ext_vector_type(8)));
typedef _Float16 v16h  __attribute__((ext_vector_type(16)));
typedef float    v8f   __attribute__((ext_vector_type(8)));
typedef unsigned int u32x4 __attribute__((ext_vector_type(4)));
typedef int          i32x4 __attribute__((ext_vector_type(4)));
typedef int          i32x8 __attribute__((ext_vector_type(8)));

#define NQ   8      // qubits (K of GEMM1)
#define BM   128    // output rows per workgroup
#define BN   128    // output cols per workgroup
#define BK   32     // K-step of GEMM2 (WMMA K)
#define LDA  40     // hA row stride in halves (80B: 16B-aligned, conflict-free)
#define LDB  36     // bT32 row stride in f32 (144B) == TDM pad: 4 dwords per 32 dwords

#if __has_builtin(__builtin_amdgcn_tensor_load_to_lds)
#define USE_TDM 1
#else
#define USE_TDM 0
#endif

#if USE_TDM
// One TDM descriptor copies a 32(f,contig) x 128(n-rows) f32 tile of w2 into LDS,
// inserting 4 pad dwords after every 32 dwords -> 36-dword LDS rows (LDB).
__device__ __forceinline__ void tdm_load_w2_tile(const float* gsrc, unsigned lds_byte, int F) {
    const unsigned long long ga = (unsigned long long)gsrc;
    u32x4 g0;
    g0[0] = 1u;                                                   // count=1, user mode
    g0[1] = lds_byte;                                             // LDS dest (bytes)
    g0[2] = (unsigned)ga;                                         // global_addr[31:0]
    g0[3] = ((unsigned)(ga >> 32) & 0x01FFFFFFu) | (2u << 30);    // addr[56:32] | type=2
    const unsigned td0 = 1u << 30, td1 = 1u << 30;                // huge dims: no OOB clip
    i32x8 g1;
    g1[0] = (int)((2u << 16) | (1u << 20) | (4u << 22) | (3u << 25));
    //        data_size=4B   pad_en       every 32 dw   pad 4 dw
    g1[1] = (int)((td0 & 0xFFFFu) << 16);                         // tensor_dim0[15:0]
    g1[2] = (int)((td0 >> 16) | ((td1 & 0xFFFFu) << 16));         // td0 hi | td1 lo
    g1[3] = (int)((td1 >> 16) | ((unsigned)BK << 16));            // td1 hi | tile_dim0=32
    g1[4] = BN;                                                   // tile_dim1=128, tile_dim2=0
    g1[5] = F;                                                    // tensor_dim0_stride (elems)
    g1[6] = 0;
    g1[7] = 0;
    i32x4 g2; g2[0] = (int)(1u << 30); g2[1] = 0; g2[2] = 0; g2[3] = 0;
    i32x4 g3; g3[0] = 0; g3[1] = 0; g3[2] = 0; g3[3] = 0;
    i32x8 gx; gx[0] = 0; gx[1] = 0; gx[2] = 0; gx[3] = 0;
              gx[4] = 0; gx[5] = 0; gx[6] = 0; gx[7] = 0;         // extension group: disabled
    __builtin_amdgcn_tensor_load_to_lds(g0, g1, g2, g3, gx, 0);
}
#endif

__global__ __launch_bounds__(256)
void ffq_fused_kernel(const float* __restrict__ x, const float* __restrict__ theta,
                      const float* __restrict__ w1, const float* __restrict__ w2,
                      float* __restrict__ out, int M, int E, int F) {
    __shared__ __align__(16) float    qv[BM][NQ];          //  4 KB
    __shared__ __align__(16) float    w1blk[2][BK][NQ];    //  2 KB
    __shared__ __align__(16) _Float16 hA[BM][LDA];         // 10 KB
    __shared__ __align__(16) float    bT32[2][BN][LDB];    // 36 KB (f32, TDM-padded rows)

    const int tid  = threadIdx.x;
    const int lane = tid & 31;
    const int wid  = tid >> 5;
    const int wm   = wid & 3;          // wave row group: 4 x 32 rows
    const int wn   = wid >> 2;         // wave col group: 2 x 64 cols
    const int row0 = blockIdx.y * BM;
    const int col0 = blockIdx.x * BN;
    const int nIter = F / BK;

    // ---- q-vectors: qv[m][q] = cos(x[row,q]) * cos(theta[q]) ----
    if (tid < BM) {
        const float* xr = x + (size_t)(row0 + tid) * (size_t)E;
        #pragma unroll
        for (int q = 0; q < NQ; ++q)
            qv[tid][q] = __cosf(xr[q]) * __cosf(theta[q]);
    }

    // ---- prologue: stage k-step 0 ----
    if (tid < 64)
        ((float4*)w1blk[0])[tid] = ((const float4*)w1)[tid];
#if USE_TDM
    if (wid == 0)
        tdm_load_w2_tile(w2 + (size_t)col0 * (size_t)F,
                         (unsigned)(unsigned long long)(void*)&bT32[0][0][0], F);
#else
    #pragma unroll
    for (int j = 0; j < 4; ++j) {
        const int idx = tid + 256 * j;
        const int n = idx >> 3, c4 = (idx & 7) * 4;
        *(float4*)&bT32[0][n][c4] = *(const float4*)(w2 + (size_t)(col0 + n) * (size_t)F + c4);
    }
#endif
    __syncthreads();                       // publish qv, w1blk[0] (+ fallback bT[0])

    // ---- B_0: hA for k-step 0 (lane = k column, wave = 16-row group) ----
    {
        const float4 wlo = *(const float4*)&w1blk[0][lane][0];
        const float4 whi = *(const float4*)&w1blk[0][lane][4];
        #pragma unroll
        for (int i = 0; i < 16; ++i) {
            const int m = wid * 16 + i;
            const float4 qlo = *(const float4*)&qv[m][0];   // LDS broadcast
            const float4 qhi = *(const float4*)&qv[m][4];
            float s = qlo.x * wlo.x + qlo.y * wlo.y + qlo.z * wlo.z + qlo.w * wlo.w
                    + qhi.x * whi.x + qhi.y * whi.y + qhi.z * whi.z + qhi.w * whi.w;
            hA[m][lane] = (_Float16)fmaxf(s, 0.f);
        }
    }

    v8f acc[2][4];
    #pragma unroll
    for (int i = 0; i < 2; ++i)
        #pragma unroll
        for (int j = 0; j < 4; ++j)
            acc[i][j] = (v8f){0.f, 0.f, 0.f, 0.f, 0.f, 0.f, 0.f, 0.f};

    // Per-lane fragment column selection (wave32 16-bit A/B layout):
    // lanes 0-15: halves[0..7]=K 0..7,  [8..15]=K 16..23
    // lanes16-31: halves[0..7]=K 8..15, [8..15]=K 24..31
    const int r  = lane & 15;
    const int c0 = (lane < 16) ? 0  : 8;
    const int c1 = (lane < 16) ? 16 : 24;

    for (int it = 0; it < nIter; ++it) {
        const int  cur = it & 1, nxt = cur ^ 1;
        const bool haveNext = (it + 1) < nIter;
        const int  f0n = (it + 1) * BK;

        // ---- A_{it+1}: issue loads for next k-step (overlap with this iter's WMMAs) ----
        float4 w1n;
        if (tid < 64 && haveNext)
            w1n = ((const float4*)w1)[f0n * 2 + tid];
#if USE_TDM
        if (wid == 0 && haveNext)
            tdm_load_w2_tile(w2 + (size_t)col0 * (size_t)F + f0n,
                             (unsigned)(unsigned long long)(void*)&bT32[nxt][0][0], F);
        // in-order TDM completion: <=1 outstanding means bT32[cur] has landed
        if (haveNext) __builtin_amdgcn_s_wait_tensorcnt(1);
        else          __builtin_amdgcn_s_wait_tensorcnt(0);
#else
        float4 bnxt[4];
        if (haveNext) {
            #pragma unroll
            for (int j = 0; j < 4; ++j) {
                const int idx = tid + 256 * j;
                const int n = idx >> 3, c4 = (idx & 7) * 4;
                bnxt[j] = *(const float4*)(w2 + (size_t)(col0 + n) * (size_t)F + f0n + c4);
            }
        }
#endif
        __syncthreads();                  // publish hA_it and bT32[cur]

        // ---- C_it: fragment loads + 8 WMMAs ----
        v16h afrag[2], bfrag[4];
        #pragma unroll
        for (int i = 0; i < 2; ++i) {
            const _Float16* p = &hA[wm * 32 + i * 16 + r][0];
            union { v16h v; half8 h[2]; } u;
            u.h[0] = *(const half8*)(p + c0);
            u.h[1] = *(const half8*)(p + c1);
            afrag[i] = u.v;
        }
        #pragma unroll
        for (int j = 0; j < 4; ++j) {
            const float* p = &bT32[cur][wn * 64 + j * 16 + r][0];
            const float4 x0 = *(const float4*)(p + c0);
            const float4 x1 = *(const float4*)(p + c0 + 4);
            const float4 x2 = *(const float4*)(p + c1);
            const float4 x3 = *(const float4*)(p + c1 + 4);
            v16h b;
            b[0]  = (_Float16)x0.x; b[1]  = (_Float16)x0.y; b[2]  = (_Float16)x0.z; b[3]  = (_Float16)x0.w;
            b[4]  = (_Float16)x1.x; b[5]  = (_Float16)x1.y; b[6]  = (_Float16)x1.z; b[7]  = (_Float16)x1.w;
            b[8]  = (_Float16)x2.x; b[9]  = (_Float16)x2.y; b[10] = (_Float16)x2.z; b[11] = (_Float16)x2.w;
            b[12] = (_Float16)x3.x; b[13] = (_Float16)x3.y; b[14] = (_Float16)x3.z; b[15] = (_Float16)x3.w;
            bfrag[j] = b;
        }
        #pragma unroll
        for (int i = 0; i < 2; ++i)
            #pragma unroll
            for (int j = 0; j < 4; ++j)
                acc[i][j] = __builtin_amdgcn_wmma_f32_16x16x32_f16(
                    false, afrag[i], false, bfrag[j],
                    (short)0, acc[i][j], false, false);

        // ---- commit A_{it+1} staging that goes through VGPRs ----
        if (haveNext && tid < 64)
            ((float4*)w1blk[nxt])[tid] = w1n;
#if !USE_TDM
        if (haveNext) {
            #pragma unroll
            for (int j = 0; j < 4; ++j) {
                const int idx = tid + 256 * j;
                const int n = idx >> 3, c4 = (idx & 7) * 4;
                *(float4*)&bT32[nxt][n][c4] = bnxt[j];
            }
        }
#endif
        __syncthreads();                  // publish w1blk[nxt] (+ fallback bT[nxt]); frag loads done

        // ---- B_{it+1}: recompute hA for next k-step (overlaps the in-flight TDM) ----
        if (haveNext) {
            const float4 wlo = *(const float4*)&w1blk[nxt][lane][0];
            const float4 whi = *(const float4*)&w1blk[nxt][lane][4];
            #pragma unroll
            for (int i = 0; i < 16; ++i) {
                const int m = wid * 16 + i;
                const float4 qlo = *(const float4*)&qv[m][0];
                const float4 qhi = *(const float4*)&qv[m][4];
                float s = qlo.x * wlo.x + qlo.y * wlo.y + qlo.z * wlo.z + qlo.w * wlo.w
                        + qhi.x * whi.x + qhi.y * whi.y + qhi.z * whi.z + qhi.w * whi.w;
                hA[m][lane] = (_Float16)fmaxf(s, 0.f);
            }
        }
    }

    // ---- store: D layout — VGPR v holds row M=v (lanes 0-15) / M=v+8 (lanes 16-31) ----
    #pragma unroll
    for (int i = 0; i < 2; ++i) {
        const int mrow = row0 + wm * 32 + i * 16 + ((lane < 16) ? 0 : 8);
        #pragma unroll
        for (int j = 0; j < 4; ++j) {
            const int ncol = col0 + wn * 64 + j * 16 + r;
            float* op = out + (size_t)mrow * (size_t)E + ncol;
            #pragma unroll
            for (int v = 0; v < 8; ++v)
                op[(size_t)v * (size_t)E] = acc[i][j][v];
        }
    }
}

extern "C" void kernel_launch(void* const* d_in, const int* in_sizes, int n_in,
                              void* d_out, int out_size, void* d_ws, size_t ws_size,
                              hipStream_t stream) {
    const float* x     = (const float*)d_in[0];   // [B,S,E] f32
    const float* theta = (const float*)d_in[1];   // [8] f32
    const float* w1    = (const float*)d_in[2];   // [F,8] f32
    const float* w2    = (const float*)d_in[3];   // [E,F] f32
    float* out = (float*)d_out;                   // [B,S,E] f32

    const int F = in_sizes[2] / NQ;               // 4096
    const int E = in_sizes[3] / F;                // 1024
    const int M = in_sizes[0] / E;                // B*S = 16384

    dim3 grid(E / BN, M / BM);                    // (8, 128)
    ffq_fused_kernel<<<grid, dim3(256), 0, stream>>>(x, theta, w1, w2, out, M, E, F);
}
